// SparseMoE_87402584473690
// MI455X (gfx1250) — matrix operand order repaired and checked by reading, not compiled, not run on hardware
//
#include <hip/hip_runtime.h>
#include <hip/hip_bf16.h>
#include <math.h>

// ---------------- problem constants (match reference setup_inputs) ----------
#define T_TOK   16384      // B*S = 4*4096
#define DDIM    1024
#define HDIM    2048
#define NEXP    8
#define TILE_M  32         // tokens per expert block (2 WMMA M-tiles)
#define CHUNK   128        // H chunk per phase-1/phase-2 round

#define KB_GU   (DDIM / 32)   // 32  K-blocks for w_gate/w_up  (K = D)
#define NB_GU   (HDIM / 16)   // 128 N-blocks for w_gate/w_up  (N = H)
#define KB_D    (HDIM / 32)   // 64  K-blocks for w_down       (K = H)
#define NB_D    (DDIM / 16)   // 64  N-blocks for w_down       (N = D)
#define FRAG_DW 256           // dwords per packed 32x16 B block (32 lanes * 8)

// dynamic LDS layout (bytes)
#define XTP_BYTES  (2 * 32 * 1024)   // 2 M-tiles * 32 K-steps * 1KB A-frag
#define ACTP_BYTES (2 * 4 * 1024)    // 2 M-tiles * 4 K-steps * 1KB A-frag
#define SMEM_BYTES (XTP_BYTES + ACTP_BYTES + 256)

typedef __attribute__((ext_vector_type(16))) __bf16       v16bf;
typedef __attribute__((ext_vector_type(8)))  float        v8f;
typedef __attribute__((ext_vector_type(4)))  unsigned int v4u;

union ABFrag { unsigned int u[8]; v4u q[2]; v16bf v; };

__device__ __forceinline__ unsigned short f2bf(float f) {
    unsigned int u = __float_as_uint(f);
    u += 0x7FFFu + ((u >> 16) & 1u);          // round-to-nearest-even
    return (unsigned short)(u >> 16);
}
__device__ __forceinline__ unsigned int pack2(float lo, float hi) {
    return (unsigned int)f2bf(lo) | ((unsigned int)f2bf(hi) << 16);
}
__device__ __forceinline__ v8f wmma_bf16(v16bf a, v16bf b, v8f c) {
    return __builtin_amdgcn_wmma_f32_16x16x32_bf16(
        false, a, false, b, (short)0, c, false, false);
}
// fragment load: two 128-bit reads, lane-contiguous (32B per lane)
__device__ __forceinline__ v16bf ldfrag_g(const unsigned int* __restrict__ p, int lane) {
    ABFrag f;
    f.q[0] = *(const v4u*)(p + lane * 8);
    f.q[1] = *(const v4u*)(p + lane * 8 + 4);
    return f.v;
}
__device__ __forceinline__ v16bf ldfrag_lds(const unsigned char* p, int lane) {
    ABFrag f;
    f.q[0] = *(const v4u*)(p + lane * 32);
    f.q[1] = *(const v4u*)(p + lane * 32 + 16);
    return f.v;
}
// map element K-offset (0..31) inside a 16-bit A/B fragment -> (lane-half, dword, halfword)
__device__ __forceinline__ unsigned fragByteOff(int mrow, int krem) {
    int kr   = krem & 15;
    int hi   = (kr >= 8) ? 1 : 0;
    int j    = ((kr & 7) >> 1) + ((krem & 16) ? 4 : 0);
    int lane = mrow + (hi << 4);
    return (unsigned)(lane * 32 + j * 4 + (krem & 1) * 2);
}

// ---------------------------------------------------------------------------
__global__ void moe_zero(float* __restrict__ out, int n, int* __restrict__ cnt) {
    if (blockIdx.x == 0 && threadIdx.x < NEXP) cnt[threadIdx.x] = 0;
    for (int i = blockIdx.x * blockDim.x + threadIdx.x; i < n;
         i += gridDim.x * blockDim.x)
        out[i] = 0.0f;
}

// Repack one fp32 weight tensor [E, Krows, Ncols] into bf16 WMMA-B fragments.
// One wave per 32x16 (KxN) block; output block = 256 dwords, lane-contiguous.
__global__ void moe_pack(const float* __restrict__ src, unsigned int* __restrict__ dst,
                         int Krows, int Ncols) {
    int wid  = (blockIdx.x * blockDim.x + threadIdx.x) >> 5;
    int lane = threadIdx.x & 31;
    int NB = Ncols >> 4;
    int blocksPerExpert = (Krows >> 5) * NB;
    int e = wid / blocksPerExpert;
    int b = wid - e * blocksPerExpert;
    int kb = b / NB, nb = b - kb * NB;
    const float* w = src + (size_t)e * Krows * Ncols;
    int col = nb * 16 + (lane & 15);
    int k0  = kb * 32 + ((lane >> 4) << 3);
    ABFrag f;
#pragma unroll
    for (int j = 0; j < 8; ++j) {
        int k = k0 + 2 * j + ((j >= 4) ? 8 : 0);
        f.u[j] = pack2(w[(size_t)k * Ncols + col], w[(size_t)(k + 1) * Ncols + col]);
    }
    unsigned int* d = dst + (size_t)wid * FRAG_DW + lane * 8;
    *(v4u*)d       = f.q[0];
    *(v4u*)(d + 4) = f.q[1];
}

// One wave32 per token: 8 logits, top-2 + softmax, append to expert lists.
__global__ void moe_router(const float* __restrict__ x,
                           const float* __restrict__ wr,
                           int* __restrict__ cnt, int* __restrict__ idxbuf,
                           float* __restrict__ wbuf) {
    int lane = threadIdx.x & 31;
    int wv   = threadIdx.x >> 5;
    int t    = blockIdx.x * 8 + wv;
    if (t >= T_TOK) return;
    const float* xr = x + (size_t)t * DDIM;
    float acc[NEXP];
#pragma unroll
    for (int e = 0; e < NEXP; ++e) acc[e] = 0.0f;
    for (int d = lane; d < DDIM; d += 32) {
        float xv = xr[d];
        const float* w = wr + (size_t)d * NEXP;
#pragma unroll
        for (int e = 0; e < NEXP; ++e) acc[e] += xv * w[e];
    }
#pragma unroll
    for (int e = 0; e < NEXP; ++e)
        for (int off = 16; off > 0; off >>= 1)
            acc[e] += __shfl_xor(acc[e], off, 32);
    if (lane == 0) {
        int i0 = 0; float m0 = acc[0];
        for (int e = 1; e < NEXP; ++e)
            if (acc[e] > m0) { m0 = acc[e]; i0 = e; }
        int i1 = -1; float m1 = -3.4e38f;
        for (int e = 0; e < NEXP; ++e)
            if (e != i0 && acc[e] > m1) { m1 = acc[e]; i1 = e; }
        float e1 = __expf(m1 - m0);            // m0 >= m1
        float inv = __builtin_amdgcn_rcpf(1.0f + e1);
        int pos0 = atomicAdd(&cnt[i0], 1);
        idxbuf[i0 * T_TOK + pos0] = t;  wbuf[i0 * T_TOK + pos0] = inv;
        int pos1 = atomicAdd(&cnt[i1], 1);
        idxbuf[i1 * T_TOK + pos1] = t;  wbuf[i1 * T_TOK + pos1] = e1 * inv;
    }
}

// Fused expert FFN for a tile of 32 routed tokens; 8 waves per block.
__global__ void __launch_bounds__(256)
moe_expert(const float* __restrict__ x,
           const unsigned int* __restrict__ pwg, const unsigned int* __restrict__ pwu,
           const unsigned int* __restrict__ pwd,
           const int* __restrict__ cnt, const int* __restrict__ idxbuf,
           const float* __restrict__ wbuf, float* __restrict__ out) {
    extern __shared__ __align__(16) unsigned char smem[];
    unsigned char* xtp  = smem;                         // A-frag packed x tile
    unsigned char* actp = smem + XTP_BYTES;             // A-frag packed act chunk
    int*   tok  = (int*)(smem + XTP_BYTES + ACTP_BYTES);
    float* twgt = (float*)(smem + XTP_BYTES + ACTP_BYTES + 128);

    int e    = blockIdx.y;
    int nE   = cnt[e];
    int base = blockIdx.x * TILE_M;
    if (base >= nE) return;

    const unsigned int* wg = pwg + (size_t)e * KB_GU * NB_GU * FRAG_DW;
    const unsigned int* wu = pwu + (size_t)e * KB_GU * NB_GU * FRAG_DW;
    const unsigned int* wd = pwd + (size_t)e * KB_D  * NB_D  * FRAG_DW;

    int tid  = threadIdx.x;
    int lane = tid & 31;
    int wv   = tid >> 5;                       // 8 waves

    if (tid < TILE_M) {
        int r = base + tid;
        tok[tid]  = (r < nE) ? idxbuf[e * T_TOK + r] : -1;
        twgt[tid] = (r < nE) ? wbuf[e * T_TOK + r] : 0.0f;
    }
    __syncthreads();

    // gather x rows, convert to bf16, scatter directly into A-fragment layout
    for (int i = tid; i < TILE_M * DDIM; i += 256) {
        int m = i >> 10;                 // token row in tile
        int c = i & (DDIM - 1);          // K index
        int t = tok[m];
        float v = (t >= 0) ? x[(size_t)t * DDIM + c] : 0.0f;
        unsigned off = (unsigned)((m >> 4) * (32 * 1024) + (c >> 5) * 1024)
                     + fragByteOff(m & 15, c & 31);
        *(unsigned short*)(xtp + off) = f2bf(v);
    }
    __syncthreads();

    int nlane = lane & 15;
    int mhi   = (lane >> 4) * 8;               // C-layout row offset within M-tile

    v8f cOut[2][8];
#pragma unroll
    for (int mt = 0; mt < 2; ++mt)
#pragma unroll
        for (int i = 0; i < 8; ++i)
#pragma unroll
            for (int r = 0; r < 8; ++r) cOut[mt][i][r] = 0.0f;

    for (int hc = 0; hc < HDIM; hc += CHUNK) {
        // ---- phase 1: G/U = x @ {Wg,Wu}[:, hc + wv*16 .. +16), K = DDIM ----
        v8f cG[2], cU[2];
#pragma unroll
        for (int mt = 0; mt < 2; ++mt)
#pragma unroll
            for (int r = 0; r < 8; ++r) { cG[mt][r] = 0.0f; cU[mt][r] = 0.0f; }

        int nb = (hc >> 4) + wv;               // N-block in w_gate/w_up
        for (int kb = 0; kb < KB_GU; ++kb) {
            v16bf a0 = ldfrag_lds(xtp + kb * 1024, lane);
            v16bf a1 = ldfrag_lds(xtp + 32 * 1024 + kb * 1024, lane);
            v16bf bg = ldfrag_g(wg + (size_t)(kb * NB_GU + nb) * FRAG_DW, lane);
            v16bf bu = ldfrag_g(wu + (size_t)(kb * NB_GU + nb) * FRAG_DW, lane);
            cG[0] = wmma_bf16(a0, bg, cG[0]);
            cG[1] = wmma_bf16(a1, bg, cG[1]);
            cU[0] = wmma_bf16(a0, bu, cU[0]);
            cU[1] = wmma_bf16(a1, bu, cU[1]);
        }

        // silu(g)*u -> bf16, scattered into A-fragment layout of act chunk
        int k_ch  = wv * 16 + nlane;           // chunk-local K of this element
        unsigned kpart = (unsigned)((k_ch >> 5) * 1024);
#pragma unroll
        for (int mt = 0; mt < 2; ++mt) {
#pragma unroll
            for (int r = 0; r < 8; ++r) {
                float g = cG[mt][r], u = cU[mt][r];
                float s = g * __builtin_amdgcn_rcpf(1.0f + __expf(-g));
                unsigned off = (unsigned)(mt * 4096) + kpart
                             + fragByteOff(r + mhi, k_ch & 31);
                *(unsigned short*)(actp + off) = f2bf(s * u);
            }
        }
        __syncthreads();

        // ---- phase 2: cOut += act_chunk @ Wd[hc..hc+128, wv*128 .. +128) ----
#pragma unroll
        for (int kk = 0; kk < 4; ++kk) {
            v16bf a0 = ldfrag_lds(actp + kk * 1024, lane);
            v16bf a1 = ldfrag_lds(actp + 4096 + kk * 1024, lane);
            int kbD = (hc >> 5) + kk;
#pragma unroll
            for (int nt = 0; nt < 8; ++nt) {
                v16bf b = ldfrag_g(
                    wd + (size_t)(kbD * NB_D + wv * 8 + nt) * FRAG_DW, lane);
                cOut[0][nt] = wmma_bf16(a0, b, cOut[0][nt]);
                cOut[1][nt] = wmma_bf16(a1, b, cOut[1][nt]);
            }
        }
        __syncthreads();   // actp reused next chunk
    }

    // scale by routing weight, accumulate into out (tokens hit <=2 experts)
#pragma unroll
    for (int mt = 0; mt < 2; ++mt) {
#pragma unroll
        for (int nt = 0; nt < 8; ++nt) {
            int col = wv * 128 + nt * 16 + nlane;
#pragma unroll
            for (int r = 0; r < 8; ++r) {
                int m = mt * 16 + r + mhi;
                int t = tok[m];
                if (t >= 0)
                    atomicAdd(&out[(size_t)t * DDIM + col], cOut[mt][nt][r] * twgt[m]);
            }
        }
    }
}

// ---------------------------------------------------------------------------
extern "C" void kernel_launch(void* const* d_in, const int* in_sizes, int n_in,
                              void* d_out, int out_size, void* d_ws, size_t ws_size,
                              hipStream_t stream) {
    const float* x  = (const float*)d_in[0];   // [T, D]
    const float* wr = (const float*)d_in[1];   // [D, E]
    const float* wg = (const float*)d_in[2];   // [E, D, H]
    const float* wu = (const float*)d_in[3];   // [E, D, H]
    const float* wd = (const float*)d_in[4];   // [E, H, D]
    float* out = (float*)d_out;                // [T, D]

    // workspace: counters | expert lists (1MB) | packed bf16 weights (96MB)
    char* ws = (char*)d_ws;
    int*   cnt    = (int*)ws;                              // 256 B
    int*   idxbuf = (int*)(ws + 256);                      // E*T ints
    float* wbuf   = (float*)(ws + 256 + NEXP * T_TOK * 4); // E*T floats
    const size_t PKO = 2u * 1024u * 1024u;
    const size_t PKSZ = (size_t)NEXP * DDIM * HDIM / 2;    // dwords per tensor
    unsigned int* pwg = (unsigned int*)(ws + PKO);
    unsigned int* pwu = pwg + PKSZ;
    unsigned int* pwd = pwu + PKSZ;

    moe_zero<<<4096, 256, 0, stream>>>(out, T_TOK * DDIM, cnt);
    moe_pack<<<NEXP * KB_GU * NB_GU / 8, 256, 0, stream>>>(wg, pwg, DDIM, HDIM);
    moe_pack<<<NEXP * KB_GU * NB_GU / 8, 256, 0, stream>>>(wu, pwu, DDIM, HDIM);
    moe_pack<<<NEXP * KB_D  * NB_D  / 8, 256, 0, stream>>>(wd, pwd, HDIM, DDIM);
    moe_router<<<T_TOK / 8, 256, 0, stream>>>(x, wr, cnt, idxbuf, wbuf);

    (void)hipFuncSetAttribute((const void*)moe_expert,
                              hipFuncAttributeMaxDynamicSharedMemorySize,
                              SMEM_BYTES);
    moe_expert<<<dim3(T_TOK / TILE_M, NEXP), 256, SMEM_BYTES, stream>>>(
        x, pwg, pwu, pwd, cnt, idxbuf, wbuf, out);
}
